// FusionTransModel_61899068670229
// MI455X (gfx1250) — compile-verified, matrix operand order checked by reading, not run on hardware
//
#include <hip/hip_runtime.h>
#include <hip/hip_bf16.h>

// ---------------------------------------------------------------------------
// CDNA5 (gfx1250) implementation of the FusionTransModel forward pass.
// wave32, WMMA 16x16x32 f16->f32 for all GEMM-shaped work, flash attention,
// TDM (tensor_load_to_lds) double-buffered LDS staging for all GEMM A-tiles.
// conv2 is folded into the unified GEMM as a K=384 overlapping-row GEMM.
// ---------------------------------------------------------------------------

typedef __attribute__((ext_vector_type(16))) _Float16 v16h;
typedef __attribute__((ext_vector_type(8)))  _Float16 v8h;
typedef __attribute__((ext_vector_type(8)))  float    v8f;
typedef unsigned int u32x4 __attribute__((ext_vector_type(4)));
typedef int          i32x4 __attribute__((ext_vector_type(4)));
typedef int          i32x8 __attribute__((ext_vector_type(8)));

#if __has_builtin(__builtin_amdgcn_tensor_load_to_lds) && __has_builtin(__builtin_amdgcn_s_wait_tensorcnt)
#define CDNA5_HAS_TDM 1
#else
#define CDNA5_HAS_TDM 0
#endif

union Frag16 { v16h v; v8h h[2]; };

#define EPI_GELU 1
#define EPI_F16  2
#define EPI_RES  4
#define EPI_POS  8
#define EPI_QKV  16
#define EPI_CONV 32

__device__ __forceinline__ float gelu_f(float x) {
    return 0.5f * x * (1.0f + erff(x * 0.7071067811865476f));
}
__device__ __forceinline__ float sigmoid_f(float x) {
    return 1.0f / (1.0f + expf(-x));
}

// Load a 16x32 f16 fragment (A-layout; identical indexing serves B when the
// source is stored N-major x K-contiguous, i.e. W[n,k] row-major).
// lane holds row (lane&15); k pattern: kb=(lane>>4)*8 -> halves {kb..kb+7, kb+16..kb+23}
__device__ __forceinline__ v16h load_frag(const _Float16* p, int ld) {
    int lane = threadIdx.x & 31;
    const _Float16* r = p + (size_t)(lane & 15) * ld + ((lane >> 4) << 3);
    Frag16 f;
    f.h[0] = *(const v8h*)(r);
    f.h[1] = *(const v8h*)(r + 16);
    return f.v;
}

__device__ __forceinline__ v8f wmma16(v16h a, v16h b, v8f c) {
    return __builtin_amdgcn_wmma_f32_16x16x32_f16(
        /*neg_a=*/false, a, /*neg_b=*/false, b,
        /*c_mod=*/(short)0, c, /*reuse_a=*/false, /*reuse_b=*/false);
}

// ---------------------------------------------------------------------------
// Small utility kernels
// ---------------------------------------------------------------------------
__global__ void cvt_f16_k(const float* __restrict__ in, _Float16* __restrict__ out, int n) {
    int i = blockIdx.x * blockDim.x + threadIdx.x;
    if (i < n) out[i] = (_Float16)in[i];
}

// w2 [E,E,3] fp32 -> [e][dl*128+k] f16  (conv2-as-GEMM weight layout, K=384)
__global__ void repack_conv2_k(const float* __restrict__ w, _Float16* __restrict__ out) {
    int i = blockIdx.x * blockDim.x + threadIdx.x;
    if (i >= 128 * 384) return;
    int e  = i / 384;
    int r  = i % 384;
    int dl = r >> 7;
    int k  = r & 127;
    out[i] = (_Float16)w[(e * 128 + k) * 3 + dl];
}

// quaternion normalize + sequential sign-continuity scan (one thread per batch)
__global__ void quat_fix_k(const float* __restrict__ qin, float* __restrict__ qout, int B, int L) {
    int b = blockIdx.x * blockDim.x + threadIdx.x;
    if (b >= B) return;
    float s = 1.0f;
    float pq0 = 0.f, pq1 = 0.f, pq2 = 0.f, pq3 = 0.f;
    for (int l = 0; l < L; ++l) {
        const float* p = qin + ((size_t)b * L + l) * 4;
        float q0 = p[0], q1 = p[1], q2 = p[2], q3 = p[3];
        float nrm = sqrtf(q0 * q0 + q1 * q1 + q2 * q2 + q3 * q3);
        float d2 = (nrm > 1e-8f) ? nrm : 1.0f;
        q0 /= d2; q1 /= d2; q2 /= d2; q3 /= d2;
        if (l > 0) {
            float d = pq0 * q0 + pq1 * q1 + pq2 * q2 + pq3 * q3;
            s = (d > 0.f) ? 1.0f : ((d < 0.f) ? -1.0f : s);
        }
        pq0 = q0; pq1 = q1; pq2 = q2; pq3 = q3;
        float* o = qout + ((size_t)b * L + l) * 4;
        o[0] = q0 * s; o[1] = q1 * s; o[2] = q2 * s; o[3] = q3 * s;
    }
}

// conv1 (k=3, same-pad) + BN(eval) + exact GELU, writes f16 padded [B, L+2, 128]
__global__ void conv1_bn_gelu_k(const float* __restrict__ x, int Cin,
                                const float* __restrict__ w1, const float* __restrict__ b1,
                                const float* __restrict__ g1, const float* __restrict__ be1,
                                _Float16* __restrict__ hpad, int L, int total) {
    int idx = blockIdx.x * blockDim.x + threadIdx.x;
    if (idx >= total) return;
    int e  = idx & 127;
    int ml = idx >> 7;
    int b  = ml / L, l = ml % L;
    float acc = b1[e];
    for (int dl = 0; dl < 3; ++dl) {
        int ll = l + dl - 1;
        if (ll < 0 || ll >= L) continue;
        const float* xp = x + ((size_t)b * L + ll) * Cin;
        const float* wp = w1 + (size_t)e * Cin * 3 + dl;
        for (int ci = 0; ci < Cin; ++ci) acc += wp[ci * 3] * xp[ci];
    }
    float inv = rsqrtf(1.0f + 1e-5f);
    acc = g1[e] * acc * inv + be1[e];
    acc = gelu_f(acc);
    hpad[((size_t)b * (L + 2) + l + 1) * 128 + e] = (_Float16)acc;
}

// ---------------------------------------------------------------------------
// Unified block-tiled WMMA GEMM: C[M,N] = A[M,K] @ W[N,K]^T
//   - 256-thread block (8 waves); block tile = 128 rows x 64 cols
//   - A tile (128x32 f16) double-buffered in LDS, staged by the Tensor Data
//     Mover (tensor_load_to_lds, TENSORcnt) when available, else manual copy
//   - lda may differ from K (overlapping rows: conv2-as-GEMM), apad = extra
//     pad rows per batch in the A row indexing
//   - epilogues: bias / pos-add / residual / GELU / f16-out / QKV scatter /
//     conv BN+GELU+weighted-fusion accumulate
// ---------------------------------------------------------------------------
struct GemmParams {
    const _Float16* A;
    const _Float16* W;
    const float* bias;
    const float* pos;
    float* Cf;
    _Float16* Ch;
    float* resid;
    _Float16* qo; _Float16* ko; _Float16* vo;     // QKV scatter
    const float* g2; const float* be2;            // conv BN affine
    const float* aw; const float* gw; const float* qw; // fusion scalars
    int M, N, K, lda, apad, Lmod, epi, sensor;
};

__global__ void gemm_block_k(GemmParams p) {
    __shared__ __align__(16) _Float16 Atile[2][128 * 32];
    int tid  = threadIdx.x;
    int wave = tid >> 5;
    int lane = tid & 31;
    int mblk = blockIdx.x * 128;
    int n0   = blockIdx.y * 64;
    int nk   = p.K >> 5;
    int baseRow = mblk + p.apad * (mblk / p.Lmod);
    const _Float16* Abase = p.A + (size_t)baseRow * p.lda;

    auto stage = [&](int buf, int ks) {
#if CDNA5_HAS_TDM
        if (wave == 0) {
            // 2D TDM descriptor: 128 rows x 16 dwords (64B), row stride lda/2 dwords
            unsigned lds = (unsigned)(size_t)(void*)&Atile[buf][0];
            unsigned long long ga =
                (unsigned long long)(size_t)(const void*)(Abase + (ks << 5));
            u32x4 g0;
            g0[0] = 1u;                                   // count=1, user mode
            g0[1] = lds;                                  // lds_addr
            g0[2] = (unsigned)(ga & 0xFFFFFFFFull);       // global_addr[31:0]
            g0[3] = (unsigned)((ga >> 32) & 0x01FFFFFFull) | (2u << 30); // addr[56:32] | type=2
            unsigned dim0 = 16u;                          // dwords per row
            unsigned dim1 = 128u;                         // rows
            unsigned long long str0 = (unsigned long long)(p.lda >> 1); // dwords
            i32x8 g1;
            g1[0] = (int)(2u << 16);                      // data_size = 4B
            g1[1] = (int)((dim0 & 0xFFFFu) << 16);        // tensor_dim0[15:0] @63:48
            g1[2] = (int)((dim0 >> 16) | ((dim1 & 0xFFFFu) << 16)); // dim0 hi | dim1 lo
            g1[3] = (int)((dim1 >> 16) | (16u << 16));    // dim1 hi | tile_dim0=16
            g1[4] = (int)128;                             // tile_dim1=128, tile_dim2=0
            g1[5] = (int)(str0 & 0xFFFFFFFFull);          // tensor_dim0_stride lo
            g1[6] = (int)((str0 >> 32) & 0xFFFFull);      // stride hi | dim1_stride lo(0)
            g1[7] = 0;
            i32x4 gz = {0, 0, 0, 0};
#if defined(__clang_major__) && __clang_major__ >= 23
            i32x8 gz8 = {0, 0, 0, 0, 0, 0, 0, 0};
            __builtin_amdgcn_tensor_load_to_lds(g0, g1, gz, gz, gz8, 0);
#else
            __builtin_amdgcn_tensor_load_to_lds(g0, g1, gz, gz, 0);
#endif
        }
#else
        // manual staging fallback: 512 x 16B chunks, 2 per thread
        for (int c = tid; c < 512; c += 256) {
            int row = c >> 2, part = c & 3;
            *(v8h*)&Atile[buf][row * 32 + part * 8] =
                *(const v8h*)(Abase + (size_t)row * p.lda + (ks << 5) + part * 8);
        }
#endif
    };

    v8f zero = {};
    v8f acc[4] = {zero, zero, zero, zero};

    stage(0, 0);
    for (int ks = 0; ks < nk; ++ks) {
#if CDNA5_HAS_TDM
        if (wave == 0) __builtin_amdgcn_s_wait_tensorcnt(0);
#endif
        __syncthreads();
        if (ks + 1 < nk) stage((ks + 1) & 1, ks + 1);

        const _Float16* at = &Atile[ks & 1][(wave << 4) * 32];
        v16h a = load_frag(at, 32);
        v16h bfr[4];
#pragma unroll
        for (int t = 0; t < 4; ++t)
            bfr[t] = load_frag(p.W + (size_t)(n0 + t * 16) * p.K + (ks << 5), p.K);
#pragma unroll
        for (int t = 0; t < 4; ++t)
            acc[t] = wmma16(a, bfr[t], acc[t]);
    }

    float inv = rsqrtf(1.0f + 1e-5f);
    float wsel = 0.0f;
    if (p.epi & EPI_CONV) {
        float wa = sigmoid_f(p.aw[0]);
        float wg = sigmoid_f(p.gw[0]);
        float wq = sigmoid_f(p.qw[0]);
        wsel = ((p.sensor == 0) ? wa : (p.sensor == 1) ? wg : wq) / (wa + wg + wq);
    }

    int m0 = mblk + (wave << 4);
    int nn = lane & 15;
    int mh = (lane >> 4) * 8;
#pragma unroll
    for (int t = 0; t < 4; ++t) {
#pragma unroll
        for (int i = 0; i < 8; ++i) {
            int m = m0 + mh + i;
            int n = n0 + t * 16 + nn;
            float v = acc[t][i];
            if (p.bias) v += p.bias[n];
            if (p.epi & EPI_QKV) {
                int which = n >> 7;
                int r = n & 127;
                int h = r >> 5;
                int d = r & 31;
                int b = m / p.Lmod, l = m % p.Lmod;
                _Float16 hv = (_Float16)v;
                if (which == 0)      p.qo[(((size_t)b * 4 + h) * p.Lmod + l) * 32 + d] = hv;
                else if (which == 1) p.ko[(((size_t)b * 4 + h) * p.Lmod + l) * 32 + d] = hv;
                else                 p.vo[(((size_t)b * 4 + h) * 32 + d) * p.Lmod + l] = hv;
                continue;
            }
            if (p.epi & EPI_CONV) {
                v = p.g2[n] * v * inv + p.be2[n];
                v = gelu_f(v) * wsel;
            }
            if (p.epi & EPI_POS) v += p.pos[(size_t)(m % p.Lmod) * p.N + n];
            if (p.epi & EPI_RES) v += p.resid[(size_t)m * p.N + n];
            if (p.epi & EPI_GELU) v = gelu_f(v);
            if (p.epi & EPI_F16) p.Ch[(size_t)m * p.N + n] = (_Float16)v;
            else                 p.Cf[(size_t)m * p.N + n] = v;
        }
    }
}

// ---------------------------------------------------------------------------
// Flash attention: one wave per (b,h,16-query tile), streaming 32 keys/iter.
// S = Q.K^T via 2 WMMAs, online softmax, P.V via 2 WMMAs (P through LDS tile).
// ---------------------------------------------------------------------------
__global__ void attn_flash_k(const _Float16* __restrict__ q, const _Float16* __restrict__ k,
                             const _Float16* __restrict__ vt, _Float16* __restrict__ o,
                             int L, int ntiles) {
    __shared__ __align__(16) _Float16 plds[4][16 * 32];
    int wave = threadIdx.x >> 5;
    int lane = threadIdx.x & 31;
    int tile = blockIdx.x * (blockDim.x >> 5) + wave;
    if (tile >= ntiles) return;
    int qtiles = L >> 4;
    int bh = tile / qtiles;
    int qt = tile % qtiles;

    v16h aQ = load_frag(q + ((size_t)bh * L + qt * 16) * 32, 32);

    v8f accO0 = {}, accO1 = {};
    float mrow[8], lrow[8];
#pragma unroll
    for (int i = 0; i < 8; ++i) { mrow[i] = -1e30f; lrow[i] = 0.0f; }

    const float scale = 0.17677669529663687f; // 1/sqrt(32)
    int nn = lane & 15;
    int mh = (lane >> 4) * 8;
    _Float16* lp = &plds[wave][0];

    for (int kc = 0; kc < L; kc += 32) {
        v16h bK0 = load_frag(k + ((size_t)bh * L + kc) * 32, 32);
        v16h bK1 = load_frag(k + ((size_t)bh * L + kc + 16) * 32, 32);
        v8f zero = {};
        v8f s0 = wmma16(aQ, bK0, zero);
        v8f s1 = wmma16(aQ, bK1, zero);
#pragma unroll
        for (int i = 0; i < 8; ++i) {
            float a0 = s0[i] * scale;
            float a1 = s1[i] * scale;
            float cm = fmaxf(a0, a1);
            for (int msk = 1; msk < 16; msk <<= 1)
                cm = fmaxf(cm, __shfl_xor(cm, msk, 32));
            float mn = fmaxf(mrow[i], cm);
            float sc = expf(mrow[i] - mn);
            float p0 = expf(a0 - mn);
            float p1 = expf(a1 - mn);
            float r = p0 + p1;
            for (int msk = 1; msk < 16; msk <<= 1)
                r += __shfl_xor(r, msk, 32);
            lrow[i] = lrow[i] * sc + r;
            mrow[i] = mn;
            accO0[i] *= sc;
            accO1[i] *= sc;
            int m = mh + i;
            lp[m * 32 + nn]      = (_Float16)p0;
            lp[m * 32 + 16 + nn] = (_Float16)p1;
        }
        v16h aP  = load_frag(lp, 32);
        v16h bV0 = load_frag(vt + ((size_t)bh * 32 + 0)  * L + kc, L);
        v16h bV1 = load_frag(vt + ((size_t)bh * 32 + 16) * L + kc, L);
        accO0 = wmma16(aP, bV0, accO0);
        accO1 = wmma16(aP, bV1, accO1);
    }

    int b = bh >> 2, h = bh & 3;
#pragma unroll
    for (int i = 0; i < 8; ++i) {
        int m = qt * 16 + mh + i;
        float inv = 1.0f / lrow[i];
        size_t base = ((size_t)b * L + m) * 128 + h * 32;
        o[base + nn]      = (_Float16)(accO0[i] * inv);
        o[base + 16 + nn] = (_Float16)(accO1[i] * inv);
    }
}

// ---------------------------------------------------------------------------
// LayerNorm over D=128, one wave per row; optional f32 and/or f16 outputs
// ---------------------------------------------------------------------------
__global__ void layernorm_k(const float* __restrict__ x, const float* __restrict__ g,
                            const float* __restrict__ b, float* __restrict__ outf,
                            _Float16* __restrict__ outh, int rows) {
    int wave = threadIdx.x >> 5;
    int lane = threadIdx.x & 31;
    int row = blockIdx.x * (blockDim.x >> 5) + wave;
    if (row >= rows) return;
    const float4* xr = (const float4*)(x + (size_t)row * 128);
    float4 v = xr[lane];
    float s  = v.x + v.y + v.z + v.w;
    float sq = v.x * v.x + v.y * v.y + v.z * v.z + v.w * v.w;
    for (int msk = 1; msk < 32; msk <<= 1) {
        s  += __shfl_xor(s,  msk, 32);
        sq += __shfl_xor(sq, msk, 32);
    }
    float mean = s * (1.0f / 128.0f);
    float var  = sq * (1.0f / 128.0f) - mean * mean;
    float rs   = rsqrtf(var + 1e-5f);
    float vals[4] = {v.x, v.y, v.z, v.w};
#pragma unroll
    for (int j = 0; j < 4; ++j) {
        int c = lane * 4 + j;
        float y = (vals[j] - mean) * rs * g[c] + b[c];
        if (outf) outf[(size_t)row * 128 + c] = y;
        if (outh) outh[(size_t)row * 128 + c] = (_Float16)y;
    }
}

// ---------------------------------------------------------------------------
// Attention pooling
// ---------------------------------------------------------------------------
__global__ void pool_logits_k(const float* __restrict__ x, const float* __restrict__ pw,
                              const float* __restrict__ pb, float* __restrict__ logits, int M) {
    int m = blockIdx.x * blockDim.x + threadIdx.x;
    if (m >= M) return;
    float v = pb[0];
    const float* xr = x + (size_t)m * 128;
    for (int c = 0; c < 128; ++c) v += xr[c] * pw[c];
    logits[m] = v;
}

__global__ void pool_reduce_k(const float* __restrict__ x, const float* __restrict__ logits,
                              float* __restrict__ pooled, int L) {
    int b = blockIdx.x;
    int tid = threadIdx.x; // 128 threads
    __shared__ float red[128];
    float mx = -1e30f;
    for (int l = tid; l < L; l += 128) mx = fmaxf(mx, logits[b * L + l]);
    red[tid] = mx;
    __syncthreads();
    for (int s = 64; s > 0; s >>= 1) {
        if (tid < s) red[tid] = fmaxf(red[tid], red[tid + s]);
        __syncthreads();
    }
    mx = red[0];
    __syncthreads();
    float se = 0.f;
    for (int l = tid; l < L; l += 128) se += expf(logits[b * L + l] - mx);
    red[tid] = se;
    __syncthreads();
    for (int s = 64; s > 0; s >>= 1) {
        if (tid < s) red[tid] += red[tid + s];
        __syncthreads();
    }
    se = red[0];
    float acc = 0.f;
    for (int l = 0; l < L; ++l) {
        float p = expf(logits[b * L + l] - mx) / se;
        acc += x[((size_t)b * L + l) * 128 + tid] * p;
    }
    pooled[b * 128 + tid] = acc;
}

// ---------------------------------------------------------------------------
// Small head FC: block per batch row, thread per output feature; optional LN+GELU
// ---------------------------------------------------------------------------
__global__ void head_fc_k(const float* __restrict__ X, const float* __restrict__ W,
                          const float* __restrict__ bias, const float* __restrict__ g,
                          const float* __restrict__ be, float* __restrict__ out,
                          int K, int N, int ln_gelu) {
    int b = blockIdx.x;
    int n = threadIdx.x;
    __shared__ float buf[128];
    float v = bias ? bias[n] : 0.0f;
    const float* xr = X + (size_t)b * K;
    const float* wr = W + (size_t)n * K;
    for (int kk = 0; kk < K; ++kk) v += xr[kk] * wr[kk];
    if (ln_gelu) {
        buf[n] = v;
        __syncthreads();
        float s = 0.f;
        for (int j = 0; j < N; ++j) s += buf[j];
        float mu = s / (float)N;
        float vs = 0.f;
        for (int j = 0; j < N; ++j) { float d = buf[j] - mu; vs += d * d; }
        float var = vs / (float)N;
        v = (v - mu) * rsqrtf(var + 1e-5f) * g[n] + be[n];
        v = gelu_f(v);
    }
    out[(size_t)b * N + n] = v;
}

// ---------------------------------------------------------------------------
// Host orchestration
// ---------------------------------------------------------------------------
extern "C" void kernel_launch(void* const* d_in, const int* in_sizes, int n_in,
                              void* d_out, int out_size, void* d_ws, size_t ws_size,
                              hipStream_t stream) {
    const int B = 128, L = 512, E = 128, M = B * L, H = 4;
    (void)in_sizes; (void)n_in; (void)out_size; (void)ws_size;

    const float* acc_x  = (const float*)d_in[0];
    const float* gyro_x = (const float*)d_in[1];
    const float* quat_x = (const float*)d_in[2];

    struct EncP { const float *w1, *b1, *g1, *be1, *w2, *b2, *g2, *be2; int cin; };
    EncP enc[3];
    for (int s = 0; s < 3; ++s) {
        int base = 3 + s * 8;
        enc[s].w1 = (const float*)d_in[base + 0];
        enc[s].b1 = (const float*)d_in[base + 1];
        enc[s].g1 = (const float*)d_in[base + 2];
        enc[s].be1 = (const float*)d_in[base + 3];
        enc[s].w2 = (const float*)d_in[base + 4];
        enc[s].b2 = (const float*)d_in[base + 5];
        enc[s].g2 = (const float*)d_in[base + 6];
        enc[s].be2 = (const float*)d_in[base + 7];
    }
    enc[0].cin = 3; enc[1].cin = 3; enc[2].cin = 4;

    const float* fw_acc  = (const float*)d_in[27];
    const float* fw_gyro = (const float*)d_in[28];
    const float* fw_quat = (const float*)d_in[29];
    const float* adapt_w = (const float*)d_in[30];
    const float* adapt_b = (const float*)d_in[31];
    const float* pos     = (const float*)d_in[32];

    struct LayP { const float *ln1_g, *ln1_b, *w_in, *b_in, *w_out, *b_out,
                              *ln2_g, *ln2_b, *ff1_w, *ff1_b, *ff2_w, *ff2_b; };
    LayP lay[2];
    for (int l = 0; l < 2; ++l) {
        int base = 33 + l * 12;
        lay[l].ln1_g = (const float*)d_in[base + 0];
        lay[l].ln1_b = (const float*)d_in[base + 1];
        lay[l].w_in  = (const float*)d_in[base + 2];
        lay[l].b_in  = (const float*)d_in[base + 3];
        lay[l].w_out = (const float*)d_in[base + 4];
        lay[l].b_out = (const float*)d_in[base + 5];
        lay[l].ln2_g = (const float*)d_in[base + 6];
        lay[l].ln2_b = (const float*)d_in[base + 7];
        lay[l].ff1_w = (const float*)d_in[base + 8];
        lay[l].ff1_b = (const float*)d_in[base + 9];
        lay[l].ff2_w = (const float*)d_in[base + 10];
        lay[l].ff2_b = (const float*)d_in[base + 11];
    }
    const float* fn_g   = (const float*)d_in[57];
    const float* fn_b   = (const float*)d_in[58];
    const float* pool_w = (const float*)d_in[59];
    const float* pool_b = (const float*)d_in[60];
    const float* c1_w = (const float*)d_in[61];
    const float* c1_b = (const float*)d_in[62];
    const float* hln1_g = (const float*)d_in[63];
    const float* hln1_b = (const float*)d_in[64];
    const float* c2_w = (const float*)d_in[65];
    const float* c2_b = (const float*)d_in[66];
    const float* hln2_g = (const float*)d_in[67];
    const float* hln2_b = (const float*)d_in[68];
    const float* c3_w = (const float*)d_in[69];
    const float* c3_b = (const float*)d_in[70];

    // --------------------- workspace carving ----------------------------
    char* ws = (char*)d_ws;
    size_t off = 0;
    auto nxt = [&](size_t bytes) -> char* {
        char* p = ws + off;
        off += (bytes + 255) & ~(size_t)255;
        return p;
    };

    _Float16* w2r16[3];
    for (int s = 0; s < 3; ++s) w2r16[s] = (_Float16*)nxt((size_t)128 * 384 * sizeof(_Float16));
    _Float16* adapt16 = (_Float16*)nxt((size_t)E * E * sizeof(_Float16));
    _Float16* win16[2]; _Float16* wout16[2]; _Float16* ff1w16[2]; _Float16* ff2w16[2];
    for (int l = 0; l < 2; ++l) {
        win16[l]  = (_Float16*)nxt((size_t)3 * E * E * sizeof(_Float16));
        wout16[l] = (_Float16*)nxt((size_t)E * E * sizeof(_Float16));
        ff1w16[l] = (_Float16*)nxt((size_t)4 * E * E * sizeof(_Float16));
        ff2w16[l] = (_Float16*)nxt((size_t)E * 4 * E * sizeof(_Float16));
    }
    float* quatN = (float*)nxt((size_t)M * 4 * sizeof(float));

    // hpad16 region, later reused as y16 (LN output f16)
    size_t hpad_bytes = (size_t)B * (L + 2) * E * sizeof(_Float16);
    char* regB = nxt(hpad_bytes);
    _Float16* hpad16 = (_Float16*)regB;
    _Float16* y16    = (_Float16*)regB;

    float* xbuf = (float*)nxt((size_t)M * E * sizeof(float)); // main activation f32

    // region C: q/k/vt/o (4 x M*E f16) == h16 (M*4E f16) == xln (M*E f32) == fused16
    size_t qslab = (size_t)M * E * sizeof(_Float16);
    char* regC = nxt(4 * qslab);
    _Float16* q16  = (_Float16*)(regC);
    _Float16* k16  = (_Float16*)(regC + qslab);
    _Float16* vt16 = (_Float16*)(regC + 2 * qslab);
    _Float16* o16  = (_Float16*)(regC + 3 * qslab);
    _Float16* h16  = (_Float16*)regC;       // M x 512 f16 (FFN hidden)
    float*    xln  = (float*)regC;          // M x 128 f32 (final LN)
    _Float16* fused16 = (_Float16*)regC;    // M x 128 f16 (adapter input)

    float* logits = (float*)nxt((size_t)M * sizeof(float));
    float* pooled = (float*)nxt((size_t)B * 128 * sizeof(float));
    float* h1     = (float*)nxt((size_t)B * 128 * sizeof(float));
    float* h2     = (float*)nxt((size_t)B * 64 * sizeof(float));

    // --------------------- weight conversion ----------------------------
    for (int s = 0; s < 3; ++s)
        repack_conv2_k<<<192, 256, 0, stream>>>(enc[s].w2, w2r16[s]);
    cvt_f16_k<<<(E * E + 255) / 256, 256, 0, stream>>>(adapt_w, adapt16, E * E);
    for (int l = 0; l < 2; ++l) {
        cvt_f16_k<<<(3 * E * E + 255) / 256, 256, 0, stream>>>(lay[l].w_in, win16[l], 3 * E * E);
        cvt_f16_k<<<(E * E + 255) / 256, 256, 0, stream>>>(lay[l].w_out, wout16[l], E * E);
        cvt_f16_k<<<(4 * E * E + 255) / 256, 256, 0, stream>>>(lay[l].ff1_w, ff1w16[l], 4 * E * E);
        cvt_f16_k<<<(4 * E * E + 255) / 256, 256, 0, stream>>>(lay[l].ff2_w, ff2w16[l], 4 * E * E);
    }

    // --------------------- quaternion preprocessing ---------------------
    quat_fix_k<<<1, 128, 0, stream>>>(quat_x, quatN, B, L);

    // --------------------- encoders + fusion ----------------------------
    hipMemsetAsync(hpad16, 0, hpad_bytes, stream);
    const float* sensor_in[3] = {acc_x, gyro_x, quatN};
    for (int s = 0; s < 3; ++s) {
        int total = M * E;
        conv1_bn_gelu_k<<<(total + 255) / 256, 256, 0, stream>>>(
            sensor_in[s], enc[s].cin, enc[s].w1, enc[s].b1, enc[s].g1, enc[s].be1,
            hpad16, L, total);
        // conv2 as K=384 overlapping-row GEMM (lda=128, apad=2 pad rows/batch)
        GemmParams cp = {};
        cp.A = hpad16; cp.W = w2r16[s]; cp.bias = enc[s].b2;
        cp.Cf = xbuf; cp.resid = xbuf;
        cp.g2 = enc[s].g2; cp.be2 = enc[s].be2;
        cp.aw = fw_acc; cp.gw = fw_gyro; cp.qw = fw_quat;
        cp.M = M; cp.N = E; cp.K = 384; cp.lda = 128; cp.apad = 2; cp.Lmod = L;
        cp.sensor = s; cp.epi = EPI_CONV | (s ? EPI_RES : 0);
        gemm_block_k<<<dim3(M / 128, E / 64), 256, 0, stream>>>(cp);
    }

    // --------------------- adapter + positional -------------------------
    cvt_f16_k<<<(M * E + 255) / 256, 256, 0, stream>>>(xbuf, fused16, M * E);
    {
        GemmParams gp = {};
        gp.A = fused16; gp.W = adapt16; gp.bias = adapt_b; gp.pos = pos;
        gp.Cf = xbuf;
        gp.M = M; gp.N = E; gp.K = E; gp.lda = E; gp.apad = 0; gp.Lmod = L;
        gp.epi = EPI_POS;
        gemm_block_k<<<dim3(M / 128, E / 64), 256, 0, stream>>>(gp);
    }

    // --------------------- transformer layers ---------------------------
    for (int l = 0; l < 2; ++l) {
        layernorm_k<<<M / 4, 128, 0, stream>>>(xbuf, lay[l].ln1_g, lay[l].ln1_b,
                                               nullptr, y16, M);
        {
            GemmParams gp = {};
            gp.A = y16; gp.W = win16[l]; gp.bias = lay[l].b_in;
            gp.qo = q16; gp.ko = k16; gp.vo = vt16;
            gp.M = M; gp.N = 384; gp.K = E; gp.lda = E; gp.apad = 0; gp.Lmod = L;
            gp.epi = EPI_QKV;
            gemm_block_k<<<dim3(M / 128, 384 / 64), 256, 0, stream>>>(gp);
        }
        {
            int ntiles = B * H * (L / 16);
            attn_flash_k<<<ntiles / 4, 128, 0, stream>>>(q16, k16, vt16, o16, L, ntiles);
        }
        {
            GemmParams gp = {};
            gp.A = o16; gp.W = wout16[l]; gp.bias = lay[l].b_out;
            gp.Cf = xbuf; gp.resid = xbuf;
            gp.M = M; gp.N = E; gp.K = E; gp.lda = E; gp.apad = 0; gp.Lmod = L;
            gp.epi = EPI_RES;
            gemm_block_k<<<dim3(M / 128, E / 64), 256, 0, stream>>>(gp);
        }
        layernorm_k<<<M / 4, 128, 0, stream>>>(xbuf, lay[l].ln2_g, lay[l].ln2_b,
                                               nullptr, y16, M);
        {
            GemmParams gp = {};
            gp.A = y16; gp.W = ff1w16[l]; gp.bias = lay[l].ff1_b;
            gp.Ch = h16;
            gp.M = M; gp.N = 512; gp.K = E; gp.lda = E; gp.apad = 0; gp.Lmod = L;
            gp.epi = EPI_GELU | EPI_F16;
            gemm_block_k<<<dim3(M / 128, 512 / 64), 256, 0, stream>>>(gp);
        }
        {
            GemmParams gp = {};
            gp.A = h16; gp.W = ff2w16[l]; gp.bias = lay[l].ff2_b;
            gp.Cf = xbuf; gp.resid = xbuf;
            gp.M = M; gp.N = E; gp.K = 512; gp.lda = 512; gp.apad = 0; gp.Lmod = L;
            gp.epi = EPI_RES;
            gemm_block_k<<<dim3(M / 128, E / 64), 256, 0, stream>>>(gp);
        }
    }

    // --------------------- final LN + pooling + head --------------------
    layernorm_k<<<M / 4, 128, 0, stream>>>(xbuf, fn_g, fn_b, xln, nullptr, M);
    pool_logits_k<<<M / 256, 256, 0, stream>>>(xln, pool_w, pool_b, logits, M);
    pool_reduce_k<<<B, 128, 0, stream>>>(xln, logits, pooled, L);

    head_fc_k<<<B, 128, 0, stream>>>(pooled, c1_w, c1_b, hln1_g, hln1_b, h1, 128, 128, 1);
    head_fc_k<<<B, 64, 0, stream>>>(h1, c2_w, c2_b, hln2_g, hln2_b, h2, 128, 64, 1);
    head_fc_k<<<B, 2, 0, stream>>>(h2, c3_w, c3_b, nullptr, nullptr, (float*)d_out, 64, 2, 0);
}
